// TritonCastleAttention_88252987998455
// MI455X (gfx1250) — matrix operand order repaired and checked by reading, not compiled, not run on hardware
//
#include <hip/hip_runtime.h>

// Fused "castle attention" for MI455X (gfx1250, wave32, WMMA).
// Per (z,h) head and 64-row i-tile: keep bf16 t1 = mask(q.vu^T) row-panel and
// la = mask(sigmoid(qu.ku^T)) panel in LDS; stream the cubic su = t1.la^T
// contraction through v_wmma_f32_16x16x32_bf16; online softmax; p.v via WMMA.
// Round 3: merged su subtile loop (shared t1 A-fragment, dual independent
// accumulators), unrolled hot loop, hoisted epilogue reciprocal.

typedef __attribute__((ext_vector_type(16))) __bf16        v16bf;
typedef __attribute__((ext_vector_type(8)))  float         v8f;
typedef __attribute__((ext_vector_type(4)))  unsigned int  u32x4;
typedef __attribute__((ext_vector_type(4)))  float         f32x4;

#define Z_   2
#define H_   8
#define N_   1024
#define D_   64
#define BM   64      // i-rows per workgroup (4 waves x 16 rows)
#define PS   1040    // LDS panel row stride (elements, 16B-aligned, bank-skewed)
#define PVS  40      // small-tile LDS row stride

union BfFrag {
  v16bf v;
  unsigned short h[16];
  __bf16 b[16];
  u32x4 q4[2];
};
union AccFrag {
  v8f v;
  float f[8];
};

// Native f32 -> bf16 (lowers to v_cvt_pk_bf16_f32 on gfx1250).
__device__ __forceinline__ unsigned short f2bf(float x) {
  union { __bf16 b; unsigned short u; } c;
  c.b = (__bf16)x;
  return c.u;
}

__device__ __forceinline__ AccFrag zeroAcc() {
  AccFrag a;
#pragma unroll
  for (int e = 0; e < 8; ++e) a.f[e] = 0.0f;
  return a;
}

__device__ __forceinline__ v8f wmma_bf16(v16bf a, v16bf b, v8f c) {
  // v_wmma_f32_16x16x32_bf16: (neg_a, A, neg_b, B, c_mod, C, reuse_a, reuse_b)
  return __builtin_amdgcn_wmma_f32_16x16x32_bf16(false, a, false, b, (short)0, c,
                                                 false, false);
}

// A-matrix fragment (16x32 bf16) from row-major f32 [rows][64], rows row0..+15,
// K-chunk base kk in {0,32}. Lane m=lane&15; lanes 0-15 hold K=0..7 / 16..23,
// lanes 16-31 hold K=8..15 / 24..31 (per ISA 16-bit A 16x32 table).
__device__ __forceinline__ v16bf load_a_f32(const float* row0, int lane, int kk) {
  const int m = lane & 15;
  const int h8 = (lane >> 4) * 8;
  const float* p = row0 + (size_t)m * D_ + kk + h8;
  union { f32x4 v4[4]; float f[16]; } t;
  t.v4[0] = *(const f32x4*)(p);
  t.v4[1] = *(const f32x4*)(p + 4);
  t.v4[2] = *(const f32x4*)(p + 16);
  t.v4[3] = *(const f32x4*)(p + 20);
  BfFrag f;
#pragma unroll
  for (int e = 0; e < 16; ++e) f.b[e] = (__bf16)t.f[e];
  return f.v;
}

// B-matrix fragment (32x16 bf16), B[k][n] = M[n0row + n][kk + k], M row-major
// f32 [rows][64]. Lane n=lane&15; lanes 0-15 hold K=0..7 / 16..23, lanes
// 16-31 hold K=8..15 / 24..31.
__device__ __forceinline__ v16bf load_b_f32(const float* row0, int lane, int kk) {
  const int n = lane & 15;
  const int h8 = (lane >> 4) * 8;
  const float* p = row0 + (size_t)n * D_ + kk + h8;
  union { f32x4 v4[4]; float f[16]; } t;
  t.v4[0] = *(const f32x4*)(p);
  t.v4[1] = *(const f32x4*)(p + 4);
  t.v4[2] = *(const f32x4*)(p + 16);
  t.v4[3] = *(const f32x4*)(p + 20);
  BfFrag f;
#pragma unroll
  for (int e = 0; e < 16; ++e) f.b[e] = (__bf16)t.f[e];
  return f.v;
}

// A fragment from bf16 LDS panel: A[m][kj] = P[(row0+m)*stride + j0 + kj]
__device__ __forceinline__ v16bf lds_a(const unsigned short* P, int stride,
                                       int row0, int j0, int lane) {
  const int m = lane & 15;
  const int h8 = (lane >> 4) * 8;
  const unsigned short* p = P + (size_t)(row0 + m) * stride + j0 + h8;
  BfFrag f;
  f.q4[0] = *(const u32x4*)(p);        // K = h8 .. h8+7
  f.q4[1] = *(const u32x4*)(p + 16);   // K = h8+16 .. h8+23
  return f.v;
}

// B fragment from bf16 LDS panel: B[k][n] = P[(n0+n)*stride + j0 + k]
__device__ __forceinline__ v16bf lds_b(const unsigned short* P, int stride,
                                       int n0, int j0, int lane) {
  const int n = lane & 15;
  const int h8 = (lane >> 4) * 8;
  const unsigned short* p = P + (size_t)(n0 + n) * stride + j0 + h8;
  BfFrag f;
  f.q4[0] = *(const u32x4*)(p);
  f.q4[1] = *(const u32x4*)(p + 16);
  return f.v;
}

__global__ void __launch_bounds__(128, 1)
castle_attn_kernel(const float* __restrict__ q, const float* __restrict__ k,
                   const float* __restrict__ v, const float* __restrict__ qu,
                   const float* __restrict__ ku, const float* __restrict__ vu,
                   float* __restrict__ out) {
  __shared__ unsigned short t1s[BM * PS];        // 133,120 B  t1 panel (bf16)
  __shared__ unsigned short las[32 * PS];        //  66,560 B  la panel (bf16)
  __shared__ unsigned short vts[D_ * PVS];       //   5,120 B  v^T 32x64 tile
  __shared__ unsigned short pst[4][16 * PVS];    //   5,120 B  per-wave p staging

  const int zh = blockIdx.y;
  const int i0 = blockIdx.x * BM;
  const size_t hoff = (size_t)zh * N_ * D_;
  q += hoff; k += hoff; v += hoff; qu += hoff; ku += hoff; vu += hoff; out += hoff;

  const int tid = threadIdx.x;
  const int lane = tid & 31;
  const int w = tid >> 5;            // wave 0..3
  const int r0 = w << 4;             // this wave's local i-row base
  const int laneN = lane & 15;
  const int laneH = lane >> 4;
  const int jend = i0 + BM;          // exclusive j / k bound, multiple of 32

  // q A-fragments for this wave's 16 rows stay in registers all kernel.
  const v16bf aQ0 = load_a_f32(q + (size_t)(i0 + r0) * D_, lane, 0);
  const v16bf aQ1 = load_a_f32(q + (size_t)(i0 + r0) * D_, lane, 32);

  // ---- Phase 1: t1[i, j] = (q.vu^T) masked j<=i, bf16 panel (wave-private rows)
  for (int j0 = 0; j0 < jend; j0 += 16) {
    AccFrag acc = zeroAcc();
    acc.v = wmma_bf16(aQ0, load_b_f32(vu + (size_t)j0 * D_, lane, 0), acc.v);
    acc.v = wmma_bf16(aQ1, load_b_f32(vu + (size_t)j0 * D_, lane, 32), acc.v);
#pragma unroll
    for (int r = 0; r < 8; ++r) {
      const int il = r0 + r + 8 * laneH;        // local i row
      const int j = j0 + laneN;                 // global j
      t1s[il * PS + j] = f2bf((j <= i0 + il) ? acc.f[r] : 0.0f);
    }
  }

  float mrun[8], lrun[8];
  AccFrag o[4];
#pragma unroll
  for (int r = 0; r < 8; ++r) { mrun[r] = -3.0e38f; lrun[r] = 0.0f; }
#pragma unroll
  for (int t = 0; t < 4; ++t) o[t] = zeroAcc();

  unsigned short* pw = pst[w];

  // ---- Phase 2: stream 32 attention columns at a time with online softmax
  for (int k0c = 0; k0c < jend; k0c += 32) {
    __syncthreads();   // previous iteration done reading las/vts

    if (k0c + 32 < jend) {
      __builtin_prefetch(k + (size_t)(k0c + 32) * D_, 0, 1);
      __builtin_prefetch(v + (size_t)(k0c + 32) * D_, 0, 1);
    }

    // v^T tile: vts[d][kv] = v[k0c+kv][d] in bf16 (coalesced global reads)
    for (int e = tid; e < 32 * D_; e += 128) {
      const int kv = e >> 6, d = e & 63;
      vts[d * PVS + kv] = f2bf(v[(size_t)(k0c + kv) * D_ + d]);
    }

    // la panel: rows k0c..k0c+31, cols j in [k0c, jend), cooperative across
    // waves. qu fragments hoisted: only 4 distinct ones per k0c.
    v16bf aQu[2][2];
#pragma unroll
    for (int kt = 0; kt < 2; ++kt) {
      aQu[kt][0] = load_a_f32(qu + (size_t)(k0c + 16 * kt) * D_, lane, 0);
      aQu[kt][1] = load_a_f32(qu + (size_t)(k0c + 16 * kt) * D_, lane, 32);
    }
    const int nwork = ((jend - k0c) >> 4) << 1;
    for (int t = w; t < nwork; t += 4) {
      const int j0 = k0c + ((t >> 1) << 4);
      const int kt = t & 1;
      AccFrag a = zeroAcc();
      a.v = wmma_bf16(aQu[kt][0], load_b_f32(ku + (size_t)j0 * D_, lane, 0), a.v);
      a.v = wmma_bf16(aQu[kt][1], load_b_f32(ku + (size_t)j0 * D_, lane, 32), a.v);
#pragma unroll
      for (int r = 0; r < 8; ++r) {
        const int krl = (kt << 4) + r + 8 * laneH;   // local k row 0..31
        const int j = j0 + laneN;
        const float sg = 1.0f / (1.0f + __expf(-a.f[r]));
        las[krl * PS + j] = f2bf((j > k0c + krl) ? sg : 0.0f);
      }
    }
    __syncthreads();

    // q.k^T for both 16-col subtiles
    AccFrag aS0 = zeroAcc(), aS1 = zeroAcc();
    aS0.v = wmma_bf16(aQ0, load_b_f32(k + (size_t)k0c * D_, lane, 0), aS0.v);
    aS0.v = wmma_bf16(aQ1, load_b_f32(k + (size_t)k0c * D_, lane, 32), aS0.v);
    aS1.v = wmma_bf16(aQ0, load_b_f32(k + (size_t)(k0c + 16) * D_, lane, 0), aS1.v);
    aS1.v = wmma_bf16(aQ1, load_b_f32(k + (size_t)(k0c + 16) * D_, lane, 32), aS1.v);

    // su = t1.la^T for both subtiles; shared t1 A-fragment per j-step,
    // independent accumulators so back-to-back WMMAs pipeline.
    AccFrag aU0 = zeroAcc(), aU1 = zeroAcc();
#pragma unroll 2
    for (int j0 = k0c; j0 < jend; j0 += 32) {   // the N^3 contraction
      const v16bf aT = lds_a(t1s, PS, r0, j0, lane);
      aU0.v = wmma_bf16(aT, lds_b(las, PS, 0, j0, lane), aU0.v);
      aU1.v = wmma_bf16(aT, lds_b(las, PS, 16, j0, lane), aU1.v);
    }

    // scores = qk - silu(su), causal masked
    float s[2][8];
#pragma unroll
    for (int r = 0; r < 8; ++r) {
      const int ig = i0 + r0 + r + 8 * laneH;
      const float u0 = aU0.f[r];
      const float u1 = aU1.f[r];
      const float g0 = 1.0f / (1.0f + __expf(-u0));
      const float g1 = 1.0f / (1.0f + __expf(-u1));
      s[0][r] = (k0c + laneN <= ig) ? (aS0.f[r] - u0 * g0) : -3.0e38f;
      s[1][r] = (k0c + 16 + laneN <= ig) ? (aS1.f[r] - u1 * g1) : -3.0e38f;
    }

    // online softmax (row = (r, laneH); reduce across the 16 lanes of a half)
#pragma unroll
    for (int r = 0; r < 8; ++r) {
      float mx = fmaxf(s[0][r], s[1][r]);
#pragma unroll
      for (int off = 8; off >= 1; off >>= 1) mx = fmaxf(mx, __shfl_xor(mx, off, 32));
      const float mnew = fmaxf(mrun[r], mx);
      const float alpha = __expf(mrun[r] - mnew);
      const float p0 = __expf(s[0][r] - mnew);
      const float p1 = __expf(s[1][r] - mnew);
      float rs = p0 + p1;
#pragma unroll
      for (int off = 8; off >= 1; off >>= 1) rs += __shfl_xor(rs, off, 32);
      lrun[r] = lrun[r] * alpha + rs;
      mrun[r] = mnew;
#pragma unroll
      for (int t = 0; t < 4; ++t) o[t].f[r] *= alpha;
      const int row = r + 8 * laneH;            // stage p for A-layout reload
      pw[row * PVS + laneN] = f2bf(p0);
      pw[row * PVS + 16 + laneN] = f2bf(p1);
    }

    // out += p (16x32) @ v (32x64)
    const v16bf aP = lds_a(pw, PVS, 0, 0, lane);
#pragma unroll
    for (int t = 0; t < 4; ++t)
      o[t].v = wmma_bf16(aP, lds_b(vts, PVS, t << 4, 0, lane), o[t].v);
  }

  // epilogue: normalize and store f32 output
#pragma unroll
  for (int r = 0; r < 8; ++r) {
    const float inv = 1.0f / lrun[r];
    const int row = i0 + r0 + r + 8 * laneH;
#pragma unroll
    for (int t = 0; t < 4; ++t)
      out[(size_t)row * D_ + (t << 4) + laneN] = o[t].f[r] * inv;
  }
}

extern "C" void kernel_launch(void* const* d_in, const int* in_sizes, int n_in,
                              void* d_out, int out_size, void* d_ws, size_t ws_size,
                              hipStream_t stream) {
  (void)in_sizes; (void)n_in; (void)d_ws; (void)ws_size; (void)out_size;
  const float* q  = (const float*)d_in[0];
  const float* k  = (const float*)d_in[1];
  const float* v  = (const float*)d_in[2];
  const float* qu = (const float*)d_in[3];
  const float* ku = (const float*)d_in[4];
  const float* vu = (const float*)d_in[5];
  float* out = (float*)d_out;

  dim3 grid(N_ / BM, Z_ * H_);   // 16 i-tiles x 16 heads = 256 workgroups
  dim3 block(128);               // 4 waves (wave32)
  castle_attn_kernel<<<grid, block, 0, stream>>>(q, k, v, qu, ku, vu, out);
}
